// i3BertModel_38001870635113
// MI455X (gfx1250) — compile-verified, matrix operand order checked by reading, not run on hardware
//
#include <hip/hip_runtime.h>
#include <hip/hip_bf16.h>

// ---------------------------------------------------------------------------
// BiRWKV + BERT hybrid forward for MI455X (gfx1250).
// Compute-bound (~325 GFLOP vs ~260MB weight traffic) -> all GEMMs routed
// through v_wmma_f32_16x16x32_f16 (f16 in, f32 accumulate). Scalar parts
// (WKV scan, LN, softmax, shifts) are bandwidth-trivial helper kernels.
//
// GEMM v4: 128x64 block tile, 8 wave32s each computing 32x32 (4 WMMAs/K-step).
// A staged row-major, B staged transposed, 80-byte padded rows so every
// fragment is two aligned ds_load_b128. Register-level software pipelining
// (next K-tile's global_load_b128s in flight during the WMMA burst), and
// TRANSB is a template parameter so each instantiation has a straight-line
// staging path (no runtime branch in the K-loop).
// ---------------------------------------------------------------------------

constexpr int Bn  = 4;
constexpr int Tn  = 512;
constexpr int Vn  = 32000;
constexpr int Dn  = 768;
constexpr int NHn = 12;
constexpr int HDn = 64;
constexpr int FFn = 3072;

typedef __attribute__((ext_vector_type(16))) _Float16 v16h;
typedef __attribute__((ext_vector_type(8)))  _Float16 v8h;
typedef __attribute__((ext_vector_type(4)))  _Float16 v4h;
typedef __attribute__((ext_vector_type(8)))  float    v8f;
typedef __attribute__((ext_vector_type(4)))  float    v4f;

#define ACT_NONE    0
#define ACT_GELU    1
#define ACT_SIGMOID 2
#define ACT_RELU2   3

__device__ __forceinline__ float apply_act(float v, int act) {
  switch (act) {
    case ACT_GELU:    return 0.5f * v * (1.0f + erff(v * 0.70710678118654752f));
    case ACT_SIGMOID: return 1.0f / (1.0f + expf(-v));
    case ACT_RELU2:   { float r = fmaxf(v, 0.0f); return r * r; }
    default:          return v;
  }
}

__device__ __forceinline__ v16h frag_from(const _Float16* p) {
  // Per ISA 7.12.2: lane's 16 halves are two contiguous 8-half (16B) runs.
  v8h lo = *(const v8h*)(p);
  v8h hi = *(const v8h*)(p + 16);
  return __builtin_shufflevector(lo, hi, 0, 1, 2, 3, 4, 5, 6, 7,
                                         8, 9, 10, 11, 12, 13, 14, 15);
}

// ---------------------------------------------------------------------------
// Strided-batched WMMA GEMM: C[z] = act(scale * A[z] @ B[z] + bias) (+C)
// Hard requirements (met at every call site): M % 128 == 0, N % 64 == 0,
// K % 32 == 0, all base pointers and leading dims 16-byte aligned.
// ---------------------------------------------------------------------------
template <int TRANSB>
__global__ __launch_bounds__(256) void gemm_wmma(
    const float* __restrict__ A, long lda, long sAo, long sAi,
    const float* __restrict__ Bm, long ldb, long sBo, long sBi,
    float* __restrict__ C, long ldc, long sCo, long sCi,
    int M, int N, int K, int nInner,
    const float* __restrict__ bias, int act, int accum, float scale)
{
  const int z  = blockIdx.z;
  const int zo = z / nInner, zi = z % nInner;
  const float* Ab = A  + (long)zo * sAo + (long)zi * sAi;
  const float* Bb = Bm + (long)zo * sBo + (long)zi * sBi;
  float*       Cb = C  + (long)zo * sCo + (long)zi * sCi;

  const int bx = blockIdx.x, by = blockIdx.y;
  const int tid  = threadIdx.x;
  const int wave = tid >> 5, lane = tid & 31;
  const int wm = wave >> 1, wn = wave & 1;       // 4 x 2 wave grid
  const int half16 = lane >> 4;                  // ISA lane group
  const int l15    = lane & 15;

  // Padded row stride 40 halves = 80 bytes = 5*16B: b128-aligned fragments,
  // and the 32-lane b128 fragment read covers each bank exactly twice.
  __shared__ _Float16 Ah[128][40];               // A tile 128 x 32, row-major
  __shared__ _Float16 Bt[64][40];                // B tile stored TRANSPOSED [n][k]

  v8f c00 = {}, c01 = {}, c10 = {}, c11 = {};

  // Per-lane fragment source pointers (fixed across K steps).
  const _Float16* aptr0 = &Ah[wm * 32 +      l15][8 * half16];
  const _Float16* aptr1 = &Ah[wm * 32 + 16 + l15][8 * half16];
  const _Float16* bptr0 = &Bt[wn * 32 +      l15][8 * half16];
  const _Float16* bptr1 = &Bt[wn * 32 + 16 + l15][8 * half16];

  const float* Arow = Ab + (long)(by * 128) * lda;

  // Per-thread staging coordinates (fixed across K steps).
  int ar[4], ac4[4];                  // A: 128x32 = 1024 float4 groups
  #pragma unroll
  for (int it = 0; it < 4; ++it) {
    int g = tid + it * 256;
    ar[it] = g >> 3; ac4[it] = (g & 7) << 2;
  }
  int br[2], bc4[2];                  // B: 32x64 = 512 float4 groups
  #pragma unroll
  for (int it = 0; it < 2; ++it) {
    int g = tid + it * 256;
    if (TRANSB) { br[it] = g >> 3; bc4[it] = (g & 7) << 2; }   // [n][k4]
    else        { br[it] = g >> 4; bc4[it] = (g & 15) << 2; }  // [k][n4]
  }

  v4f pA[4], pB[2];
  auto load_tiles = [&](int k0) {
    #pragma unroll
    for (int it = 0; it < 4; ++it)
      pA[it] = *(const v4f*)(Arow + (long)ar[it] * lda + k0 + ac4[it]);
    #pragma unroll
    for (int it = 0; it < 2; ++it) {
      if (TRANSB)
        pB[it] = *(const v4f*)(Bb + (long)(bx * 64 + br[it]) * ldb + k0 + bc4[it]);
      else
        pB[it] = *(const v4f*)(Bb + (long)(k0 + br[it]) * ldb + bx * 64 + bc4[it]);
    }
  };
  auto store_tiles = [&]() {
    #pragma unroll
    for (int it = 0; it < 4; ++it)
      *(v4h*)&Ah[ar[it]][ac4[it]] = __builtin_convertvector(pA[it], v4h);
    #pragma unroll
    for (int it = 0; it < 2; ++it) {
      v4h h = __builtin_convertvector(pB[it], v4h);
      if (TRANSB) {
        *(v4h*)&Bt[br[it]][bc4[it]] = h;
      } else {
        Bt[bc4[it] + 0][br[it]] = h[0];
        Bt[bc4[it] + 1][br[it]] = h[1];
        Bt[bc4[it] + 2][br[it]] = h[2];
        Bt[bc4[it] + 3][br[it]] = h[3];
      }
    }
  };

  load_tiles(0);                                 // prologue prefetch
  for (int k0 = 0; k0 < K; k0 += 32) {
    store_tiles();                               // cvt + LDS store of tile k0
    __syncthreads();
    if (k0 + 32 < K) load_tiles(k0 + 32);        // next tile in flight during WMMAs

    v16h a0 = frag_from(aptr0);
    v16h a1 = frag_from(aptr1);
    v16h b0 = frag_from(bptr0);
    v16h b1 = frag_from(bptr1);
    c00 = __builtin_amdgcn_wmma_f32_16x16x32_f16(false, a0, false, b0, (short)0, c00, false, false);
    c01 = __builtin_amdgcn_wmma_f32_16x16x32_f16(false, a0, false, b1, (short)0, c01, false, false);
    c10 = __builtin_amdgcn_wmma_f32_16x16x32_f16(false, a1, false, b0, (short)0, c10, false, false);
    c11 = __builtin_amdgcn_wmma_f32_16x16x32_f16(false, a1, false, b1, (short)0, c11, false, false);
    __syncthreads();
  }

  // Epilogue. C/D layout: lane l15 -> column, VGPR i -> row i + 8*laneGroup.
  #pragma unroll
  for (int ti = 0; ti < 2; ++ti) {
    #pragma unroll
    for (int tj = 0; tj < 2; ++tj) {
      v8f cc = ti ? (tj ? c11 : c10) : (tj ? c01 : c00);
      int gn = bx * 64 + wn * 32 + tj * 16 + l15;
      float bv = bias ? bias[gn] : 0.0f;
      int gm0 = by * 128 + wm * 32 + ti * 16 + half16 * 8;
      #pragma unroll
      for (int i = 0; i < 8; ++i) {
        float v = cc[i] * scale + bv;
        v = apply_act(v, act);
        float* cp = Cb + (long)(gm0 + i) * ldc + gn;
        if (accum) v += *cp;
        *cp = v;
      }
    }
  }
}

// ---------------------------------------------------------------------------
// Block reductions (blockDim.x == 256)
// ---------------------------------------------------------------------------
__device__ __forceinline__ float block_sum(float v, float* sh) {
  int t = threadIdx.x;
  sh[t] = v; __syncthreads();
  #pragma unroll
  for (int s = 128; s > 0; s >>= 1) {
    if (t < s) sh[t] += sh[t + s];
    __syncthreads();
  }
  float r = sh[0]; __syncthreads();
  return r;
}
__device__ __forceinline__ float block_max(float v, float* sh) {
  int t = threadIdx.x;
  sh[t] = v; __syncthreads();
  #pragma unroll
  for (int s = 128; s > 0; s >>= 1) {
    if (t < s) sh[t] = fmaxf(sh[t], sh[t + s]);
    __syncthreads();
  }
  float r = sh[0]; __syncthreads();
  return r;
}

// Embedding gather + add + LayerNorm. One block per (b,t) row; D = 3*256.
__global__ __launch_bounds__(256) void embed_ln_k(
    const int* __restrict__ ids, const int* __restrict__ segs,
    const float* __restrict__ we, const float* __restrict__ pe,
    const float* __restrict__ te, const float* __restrict__ g,
    const float* __restrict__ b, float* __restrict__ out)
{
  __shared__ float sh[256];
  long row = blockIdx.x;                       // b*T + t
  int t  = (int)(row % Tn);
  int id = ids[row], sg = segs[row];
  float vals[3]; float s = 0.f, s2 = 0.f;
  #pragma unroll
  for (int i = 0; i < 3; ++i) {
    int d = threadIdx.x + i * 256;
    float v = we[(long)id * Dn + d] + pe[(long)t * Dn + d] + te[(long)sg * Dn + d];
    vals[i] = v; s += v; s2 += v * v;
  }
  s  = block_sum(s, sh);
  s2 = block_sum(s2, sh);
  float mean = s / Dn, var = s2 / Dn - mean * mean;
  float inv = rsqrtf(var + 1e-5f);
  float* orow = out + row * Dn;
  #pragma unroll
  for (int i = 0; i < 3; ++i) {
    int d = threadIdx.x + i * 256;
    orow[d] = (vals[i] - mean) * inv * g[d] + b[d];
  }
}

__global__ __launch_bounds__(256) void layernorm_k(
    const float* __restrict__ in, const float* __restrict__ g,
    const float* __restrict__ b, float* __restrict__ out)
{
  __shared__ float sh[256];
  long row = blockIdx.x;
  const float* xr = in + row * Dn;
  float vals[3]; float s = 0.f, s2 = 0.f;
  #pragma unroll
  for (int i = 0; i < 3; ++i) {
    int d = threadIdx.x + i * 256;
    float v = xr[d]; vals[i] = v; s += v; s2 += v * v;
  }
  s  = block_sum(s, sh);
  s2 = block_sum(s2, sh);
  float mean = s / Dn, var = s2 / Dn - mean * mean;
  float inv = rsqrtf(var + 1e-5f);
  float* orow = out + row * Dn;
  #pragma unroll
  for (int i = 0; i < 3; ++i) {
    int d = threadIdx.x + i * 256;
    orow[d] = (vals[i] - mean) * inv * g[d] + b[d];
  }
}

// Token-shift + lerp mix: out = x*m + shift(x)*(1-m). ov/mv optional.
__global__ __launch_bounds__(256) void shiftmix_k(
    const float* __restrict__ x,
    const float* __restrict__ mk, const float* __restrict__ mv,
    const float* __restrict__ mr,
    float* __restrict__ ok, float* __restrict__ ov, float* __restrict__ orr)
{
  long i = (long)blockIdx.x * 256 + threadIdx.x;
  if (i >= (long)Bn * Tn * Dn) return;
  int d = (int)(i % Dn);
  int t = (int)((i / Dn) % Tn);
  float xc = x[i];
  float xp = (t > 0) ? x[i - Dn] : 0.0f;
  float a = mk[d];  ok[i]  = xc * a + xp * (1.0f - a);
  if (ov) { float m = mv[d]; ov[i] = xc * m + xp * (1.0f - m); }
  float c = mr[d];  orr[i] = xc * c + xp * (1.0f - c);
}

// RWKV v4 WKV recurrence, max-stabilized. One thread per (batch, channel);
// channel-adjacent threads give coalesced loads each timestep.
__global__ __launch_bounds__(256) void wkv_k(
    const float* __restrict__ k, const float* __restrict__ v,
    const float* __restrict__ td, const float* __restrict__ tf,
    float* __restrict__ y)
{
  int i = blockIdx.x * 256 + threadIdx.x;
  if (i >= Bn * Dn) return;
  int b = i / Dn, c = i % Dn;
  float u = tf[c];
  float w = -expf(td[c]);
  float aa = 0.f, bb = 0.f, pp = -1e30f;
  const float* kp = k + (long)b * Tn * Dn + c;
  const float* vp = v + (long)b * Tn * Dn + c;
  float*       yp = y + (long)b * Tn * Dn + c;
  for (int t = 0; t < Tn; ++t) {
    float kt = kp[(long)t * Dn];
    float vt = vp[(long)t * Dn];
    float ww = u + pp;
    float p  = fmaxf(ww, kt);
    float e1 = expf(ww - p), e2 = expf(kt - p);
    yp[(long)t * Dn] = (aa * e1 + vt * e2) / (bb * e1 + e2 + 1e-6f);
    float ww2 = w + pp;
    float p2  = fmaxf(ww2, kt);
    float e1b = expf(ww2 - p2), e2b = expf(kt - p2);
    aa = aa * e1b + vt * e2b;
    bb = bb * e1b + e2b;
    pp = p2;
  }
}

__global__ __launch_bounds__(256) void ew_mul_k(
    const float* __restrict__ a, const float* __restrict__ b,
    float* __restrict__ o, long n)
{
  long i = (long)blockIdx.x * 256 + threadIdx.x;
  if (i < n) o[i] = a[i] * b[i];
}
__global__ __launch_bounds__(256) void ew_muladd_k(
    const float* __restrict__ a, const float* __restrict__ b,
    float* __restrict__ o, long n)
{
  long i = (long)blockIdx.x * 256 + threadIdx.x;
  if (i < n) o[i] += a[i] * b[i];
}

// Reverse along time; add=1 accumulates into out.
__global__ __launch_bounds__(256) void flip_time_k(
    const float* __restrict__ in, float* __restrict__ out, int add)
{
  long i = (long)blockIdx.x * 256 + threadIdx.x;
  if (i >= (long)Bn * Tn * Dn) return;
  int d = (int)(i % Dn);
  long bt = i / Dn;
  int t = (int)(bt % Tn);
  long b = bt / Tn;
  long src = ((b * Tn + (Tn - 1 - t)) * (long)Dn) + d;
  float v = in[src];
  if (add) out[i] += v; else out[i] = v;
}

// Masked softmax over keys. Row = (b*NH + h)*T + q, T = 2*256 per thread.
// Scores arrive already scaled by 1/sqrt(HD) (GEMM epilogue).
__global__ __launch_bounds__(256) void softmax_mask_k(
    float* __restrict__ s, const int* __restrict__ ids)
{
  __shared__ float sh[256];
  long row = blockIdx.x;
  int b = (int)(row / ((long)NHn * Tn));
  float* sp = s + row * Tn;
  const int* idb = ids + (long)b * Tn;
  float loc[2]; float mx = -3.0e38f;
  #pragma unroll
  for (int i = 0; i < 2; ++i) {
    int j = threadIdx.x + i * 256;
    float v = sp[j];
    if (idb[j] == 1) v = -1e9f;
    loc[i] = v; mx = fmaxf(mx, v);
  }
  mx = block_max(mx, sh);
  float sum = 0.f;
  #pragma unroll
  for (int i = 0; i < 2; ++i) { float e = expf(loc[i] - mx); loc[i] = e; sum += e; }
  sum = block_sum(sum, sh);
  float inv = 1.0f / sum;
  #pragma unroll
  for (int i = 0; i < 2; ++i) sp[threadIdx.x + i * 256] = loc[i] * inv;
}

// pooled[b] = tanh(x[b,0] @ pool_W + pool_b) ; one block per batch.
__global__ __launch_bounds__(256) void pool_k(
    const float* __restrict__ x, const float* __restrict__ W,
    const float* __restrict__ bias, float* __restrict__ pooled)
{
  int b = blockIdx.x;
  const float* xr = x + (long)b * Tn * Dn;   // t = 0 row
  #pragma unroll
  for (int jj = 0; jj < 3; ++jj) {
    int j = threadIdx.x + jj * 256;
    float acc = bias[j];
    for (int d = 0; d < Dn; ++d) acc += xr[d] * W[(long)d * Dn + j];
    pooled[(long)b * Dn + j] = tanhf(acc);
  }
}

// nsp = pooled @ nsp_W + nsp_b  -> 8 scalars.
__global__ void nsp_k(const float* __restrict__ pooled,
                      const float* __restrict__ W,
                      const float* __restrict__ bias,
                      float* __restrict__ out)
{
  int i = threadIdx.x;
  if (i >= Bn * 2) return;
  int b = i >> 1, c = i & 1;
  float acc = bias[c];
  for (int j = 0; j < Dn; ++j) acc += pooled[(long)b * Dn + j] * W[j * 2 + c];
  out[i] = acc;
}

// ---------------------------------------------------------------------------
// Host orchestration
// ---------------------------------------------------------------------------
struct TMp { const float *mk, *mv, *mr, *Wk, *Wv, *Wr, *Wo, *td, *tf; };
struct CMp { const float *mk, *mr, *Wk, *Wv, *Wr; };
struct RLp { const float *ln1g, *ln1b, *ln2g, *ln2b; TMp fwd, bwd; CMp cm; };
struct ALp { const float *ln1g, *ln1b, *ln2g, *ln2b, *Wqkv, *bqkv, *Wo, *bo, *W1, *b1, *W2, *b2; };

extern "C" void kernel_launch(void* const* d_in, const int* in_sizes, int n_in,
                              void* d_out, int out_size, void* d_ws, size_t ws_size,
                              hipStream_t stream) {
  (void)in_sizes; (void)n_in; (void)out_size; (void)ws_size;

  // ---- unpack inputs in setup_inputs() dict (insertion) order --------------
  int ii = 0;
  const int* input_ids   = (const int*)d_in[ii++];
  const int* segment_ids = (const int*)d_in[ii++];
  auto F = [&]() { return (const float*)d_in[ii++]; };

  const float* word_emb = F();
  const float* pos_emb  = F();
  const float* type_emb = F();
  const float* emb_ln_g = F();
  const float* emb_ln_b = F();

  RLp rl[3];
  for (int l = 0; l < 3; ++l) {
    rl[l].ln1g = F(); rl[l].ln1b = F(); rl[l].ln2g = F(); rl[l].ln2b = F();
    rl[l].fwd = { F(), F(), F(), F(), F(), F(), F(), F(), F() };
    rl[l].bwd = { F(), F(), F(), F(), F(), F(), F(), F(), F() };
    rl[l].cm  = { F(), F(), F(), F(), F() };
  }
  ALp al[3];
  for (int l = 0; l < 3; ++l) {
    al[l] = { F(), F(), F(), F(), F(), F(), F(), F(), F(), F(), F(), F() };
  }
  const float* mlm_W1   = F();
  const float* mlm_b1   = F();
  const float* mlm_ln_g = F();
  const float* mlm_ln_b = F();
  const float* mlm_W2   = F();
  const float* mlm_b2   = F();
  const float* pool_W   = F();
  const float* pool_b   = F();
  const float* nsp_W    = F();
  const float* nsp_b    = F();

  // ---- workspace layout (~150 MB of fp32) ----------------------------------
  const long NXD = (long)Bn * Tn * Dn;          // 1,572,864
  float* ws = (float*)d_ws;
  long off = 0;
  auto alloc = [&](long n) { float* p = ws + off; off += n; return p; };
  float* x      = alloc(NXD);
  float* xn     = alloc(NXD);
  float* bufA   = alloc(NXD);
  float* bufB   = alloc(NXD);
  float* bufC   = alloc(NXD);
  float* kb     = alloc(NXD);
  float* vb     = alloc(NXD);
  float* rb     = alloc(NXD);
  float* wkvb   = alloc(NXD);
  float* gb     = alloc(NXD);
  float* big    = alloc((long)Bn * Tn * FFn);            // qkv / ffn hidden
  float* scores = alloc((long)Bn * NHn * Tn * Tn);
  float* ob     = alloc(NXD);
  float* pooled = alloc((long)Bn * Dn);

  const int rows    = Bn * Tn;                  // 2048
  const int ewBlks  = (int)((NXD + 255) / 256);
  dim3 blk(256);

  // plain (non-batched) GEMM helper; M rows per block = 128.
  auto gemm = [&](const float* A, long lda, const float* Bm, long ldb,
                  float* C, long ldc, int M, int N, int K,
                  const float* bias, int act, int accum, float scale) {
    dim3 grid(N / 64, M / 128, 1);
    gemm_wmma<0><<<grid, blk, 0, stream>>>(A, lda, 0L, 0L, Bm, ldb, 0L, 0L,
                                           C, ldc, 0L, 0L, M, N, K, 1,
                                           bias, act, accum, scale);
  };

  // ---- embeddings + LN -----------------------------------------------------
  embed_ln_k<<<rows, blk, 0, stream>>>(input_ids, segment_ids, word_emb, pos_emb,
                                       type_emb, emb_ln_g, emb_ln_b, x);

  // ---- BiRWKV layers -------------------------------------------------------
  for (int l = 0; l < 3; ++l) {
    const RLp& p = rl[l];
    layernorm_k<<<rows, blk, 0, stream>>>(x, p.ln1g, p.ln1b, xn);

    // forward-time mix
    shiftmix_k<<<ewBlks, blk, 0, stream>>>(xn, p.fwd.mk, p.fwd.mv, p.fwd.mr,
                                           bufA, bufB, bufC);
    gemm(bufA, Dn, p.fwd.Wk, Dn, kb, Dn, rows, Dn, Dn, nullptr, ACT_NONE, 0, 1.f);
    gemm(bufB, Dn, p.fwd.Wv, Dn, vb, Dn, rows, Dn, Dn, nullptr, ACT_NONE, 0, 1.f);
    gemm(bufC, Dn, p.fwd.Wr, Dn, rb, Dn, rows, Dn, Dn, nullptr, ACT_SIGMOID, 0, 1.f);
    wkv_k<<<(Bn * Dn + 255) / 256, blk, 0, stream>>>(kb, vb, p.fwd.td, p.fwd.tf, wkvb);
    ew_mul_k<<<ewBlks, blk, 0, stream>>>(rb, wkvb, gb, NXD);
    gemm(gb, Dn, p.fwd.Wo, Dn, x, Dn, rows, Dn, Dn, nullptr, ACT_NONE, 1, 1.f);

    // backward-time mix (run forward machinery on time-flipped xn)
    flip_time_k<<<ewBlks, blk, 0, stream>>>(xn, ob, 0);
    shiftmix_k<<<ewBlks, blk, 0, stream>>>(ob, p.bwd.mk, p.bwd.mv, p.bwd.mr,
                                           bufA, bufB, bufC);
    gemm(bufA, Dn, p.bwd.Wk, Dn, kb, Dn, rows, Dn, Dn, nullptr, ACT_NONE, 0, 1.f);
    gemm(bufB, Dn, p.bwd.Wv, Dn, vb, Dn, rows, Dn, Dn, nullptr, ACT_NONE, 0, 1.f);
    gemm(bufC, Dn, p.bwd.Wr, Dn, rb, Dn, rows, Dn, Dn, nullptr, ACT_SIGMOID, 0, 1.f);
    wkv_k<<<(Bn * Dn + 255) / 256, blk, 0, stream>>>(kb, vb, p.bwd.td, p.bwd.tf, wkvb);
    ew_mul_k<<<ewBlks, blk, 0, stream>>>(rb, wkvb, gb, NXD);
    gemm(gb, Dn, p.bwd.Wo, Dn, bufA, Dn, rows, Dn, Dn, nullptr, ACT_NONE, 0, 1.f);
    flip_time_k<<<ewBlks, blk, 0, stream>>>(bufA, x, 1);   // x += flip(bwd_out)

    // channel mix
    layernorm_k<<<rows, blk, 0, stream>>>(x, p.ln2g, p.ln2b, xn);
    shiftmix_k<<<ewBlks, blk, 0, stream>>>(xn, p.cm.mk, nullptr, p.cm.mr,
                                           bufA, nullptr, bufC);
    gemm(bufA, Dn, p.cm.Wk, FFn, big, FFn, rows, FFn, Dn, nullptr, ACT_RELU2, 0, 1.f);
    gemm(bufC, Dn, p.cm.Wr, Dn, rb, Dn, rows, Dn, Dn, nullptr, ACT_SIGMOID, 0, 1.f);
    gemm(big, FFn, p.cm.Wv, Dn, kb, Dn, rows, Dn, FFn, nullptr, ACT_NONE, 0, 1.f);
    ew_muladd_k<<<ewBlks, blk, 0, stream>>>(rb, kb, x, NXD);
  }

  // ---- attention layers ----------------------------------------------------
  const long qkvLD = 3L * Dn;
  for (int l = 0; l < 3; ++l) {
    const ALp& p = al[l];
    layernorm_k<<<rows, blk, 0, stream>>>(x, p.ln1g, p.ln1b, xn);
    gemm(xn, Dn, p.Wqkv, qkvLD, big, qkvLD, rows, 3 * Dn, Dn, p.bqkv, ACT_NONE, 0, 1.f);

    // scores[b,h] = (Q @ K^T) / sqrt(HD) ; batch z = b*NH + h
    {
      dim3 grid(Tn / 64, Tn / 128, Bn * NHn);
      gemm_wmma<1><<<grid, blk, 0, stream>>>(
          big,          qkvLD, (long)Tn * qkvLD, (long)HDn,   // Q
          big + Dn,     qkvLD, (long)Tn * qkvLD, (long)HDn,   // K (transB)
          scores, Tn, (long)NHn * Tn * Tn, (long)Tn * Tn,
          Tn, Tn, HDn, NHn, nullptr, ACT_NONE, 0, 0.125f);
    }
    softmax_mask_k<<<Bn * NHn * Tn, blk, 0, stream>>>(scores, input_ids);

    // O[b,h] = P @ V
    {
      dim3 grid(HDn / 64, Tn / 128, Bn * NHn);
      gemm_wmma<0><<<grid, blk, 0, stream>>>(
          scores, Tn, (long)NHn * Tn * Tn, (long)Tn * Tn,     // P
          big + 2 * Dn, qkvLD, (long)Tn * qkvLD, (long)HDn,   // V
          ob, Dn, (long)Tn * Dn, (long)HDn,
          Tn, HDn, Tn, NHn, nullptr, ACT_NONE, 0, 1.0f);
    }
    gemm(ob, Dn, p.Wo, Dn, x, Dn, rows, Dn, Dn, p.bo, ACT_NONE, 1, 1.f);

    // FFN
    layernorm_k<<<rows, blk, 0, stream>>>(x, p.ln2g, p.ln2b, xn);
    gemm(xn, Dn, p.W1, FFn, big, FFn, rows, FFn, Dn, p.b1, ACT_GELU, 0, 1.f);
    gemm(big, FFn, p.W2, Dn, x, Dn, rows, Dn, FFn, p.b2, ACT_NONE, 1, 1.f);
  }

  // ---- MLM head (dominant GEMM: 2048 x 32000 x 768) ------------------------
  gemm(x, Dn, mlm_W1, Dn, bufA, Dn, rows, Dn, Dn, mlm_b1, ACT_GELU, 0, 1.f);
  layernorm_k<<<rows, blk, 0, stream>>>(bufA, mlm_ln_g, mlm_ln_b, bufB);
  gemm(bufB, Dn, mlm_W2, Vn, (float*)d_out, Vn, rows, Vn, Dn, mlm_b2, ACT_NONE, 0, 1.f);

  // ---- pooler + NSP --------------------------------------------------------
  pool_k<<<Bn, blk, 0, stream>>>(x, pool_W, pool_b, pooled);
  nsp_k<<<1, 32, 0, stream>>>(pooled, nsp_W, nsp_b,
                              (float*)d_out + (long)Bn * Tn * Vn);
}
